// AttentionBlock_91225105367083
// MI455X (gfx1250) — matrix operand order repaired and checked by reading, not compiled, not run on hardware
//
#include <hip/hip_runtime.h>

// ---------------------------------------------------------------------------
// Complex attention block for MI455X (gfx1250, wave32, WMMA f16 path).
// Pipeline: LN -> QKV cgemm -> flash attention -> out-proj cgemm -> conv+res.
// ---------------------------------------------------------------------------

typedef _Float16 h16;
typedef __attribute__((ext_vector_type(16))) _Float16 v16h;
typedef __attribute__((ext_vector_type(8)))  float    v8f;

#define BATCH   2
#define CHN     128
#define SPATIAL 4096
#define NHEAD   4
#define DHEAD   32
#define MROWS   (BATCH * SPATIAL)                 // 8192
#define QSZ     (BATCH * NHEAD * SPATIAL * DHEAD) // 1048576 elems per q/k/v plane

__device__ __forceinline__ v16h ld_2x8(const h16* p0, const h16* p1) {
  union { v16h v; uint4 u[2]; } r;
  r.u[0] = *(const uint4*)p0;
  r.u[1] = *(const uint4*)p1;
  return r.v;
}
__device__ __forceinline__ v16h ld_16(const h16* p) { return ld_2x8(p, p + 8); }

__device__ __forceinline__ v8f wmma16(v16h a, v16h b, v8f c) {
  // D = A(16x32 f16) x B(32x16 f16) + C(16x16 f32)
  return __builtin_amdgcn_wmma_f32_16x16x32_f16(false, a, false, b, (short)0, c,
                                                false, false);
}

// ---------------------------------------------------------------------------
// Complex LayerNorm over channels. One wave per (b,s) row; lane owns 4 chans.
// Writes yn as f16 [MROWS, CHN].
// ---------------------------------------------------------------------------
__global__ __launch_bounds__(256) void ln_kernel(
    const float* __restrict__ xr, const float* __restrict__ xi,
    const float* __restrict__ gwr, const float* __restrict__ gwi,
    const float* __restrict__ gbr, const float* __restrict__ gbi,
    h16* __restrict__ ynR, h16* __restrict__ ynI) {
  int wave = threadIdx.x >> 5, lane = threadIdx.x & 31;
  int row = blockIdx.x * 8 + wave;       // 0..8191
  int bb = row >> 12, ss = row & (SPATIAL - 1);
  float vr[4], vi[4];
  float sr = 0.f, si = 0.f;
#pragma unroll
  for (int j = 0; j < 4; ++j) {
    int c = lane + 32 * j;
    size_t idx = ((size_t)(bb * CHN + c)) * SPATIAL + ss;
    vr[j] = xr[idx]; vi[j] = xi[idx];
    sr += vr[j]; si += vi[j];
  }
#pragma unroll
  for (int off = 16; off > 0; off >>= 1) {
    sr += __shfl_xor(sr, off, 32);
    si += __shfl_xor(si, off, 32);
  }
  float mur = sr * (1.0f / CHN), mui = si * (1.0f / CHN);
  float sv = 0.f;
#pragma unroll
  for (int j = 0; j < 4; ++j) {
    vr[j] -= mur; vi[j] -= mui;
    sv += vr[j] * vr[j] + vi[j] * vi[j];
  }
#pragma unroll
  for (int off = 16; off > 0; off >>= 1) sv += __shfl_xor(sv, off, 32);
  float inv = rsqrtf(sv * (1.0f / CHN) + 1e-5f);
#pragma unroll
  for (int j = 0; j < 4; ++j) {
    int c = lane + 32 * j;
    float yr = vr[j] * inv, yi = vi[j] * inv;
    float wr_ = gwr[c], wi_ = gwi[c];
    ynR[(size_t)row * CHN + c] = (h16)(yr * wr_ - yi * wi_ + gbr[c]);
    ynI[(size_t)row * CHN + c] = (h16)(yr * wi_ + yi * wr_ + gbi[c]);
  }
}

// ---------------------------------------------------------------------------
// f32 -> f16 conversion (weights).
// ---------------------------------------------------------------------------
__global__ void cvt_kernel(const float* __restrict__ src, h16* __restrict__ dst,
                           int n) {
  int i = blockIdx.x * 256 + threadIdx.x;
  if (i < n) dst[i] = (h16)src[i];
}

// ---------------------------------------------------------------------------
// Complex GEMM: out[m,n] = sum_k A[m,k] * W[n,k]  (A,W complex; K=128 fixed)
// One wave computes one 16x16 tile via 4 K-chunks x 4 WMMAs.
// mode 0: QKV  (N=384, head-split f16 stores into Q/K/V planes)
// mode 1: out-projection (N=128, row-major f16 store)
// mode 2: 1x1 conv + residual -> fp32 d_out in [2,B,C,S] layout (NT stores)
// ---------------------------------------------------------------------------
__global__ __launch_bounds__(128) void cgemm_kernel(
    const h16* __restrict__ Ar, const h16* __restrict__ Ai,
    const h16* __restrict__ Wr, const h16* __restrict__ Wi,
    const float* __restrict__ biasR, const float* __restrict__ biasI,
    int mode,
    h16* __restrict__ outR0, h16* __restrict__ outI0,
    float* __restrict__ outF,
    const float* __restrict__ xr, const float* __restrict__ xi) {
  int wave = threadIdx.x >> 5, lane = threadIdx.x & 31;
  int mbase = (blockIdx.y * 4 + wave) * 16;
  int nbase = blockIdx.x * 16;
  int col = lane & 15;
  int rowoff = (lane < 16) ? 0 : 8;   // C/D layout: hi half holds rows M+8
  int d0  = (lane < 16) ? 0 : 8;      // A layout K split
  int bko = (lane < 16) ? 0 : 8;      // B layout K split
  int mrow = mbase + col;
  int ecol = nbase + col;

  const h16* arow  = Ar + (size_t)mrow * CHN;
  const h16* airow = Ai + (size_t)mrow * CHN;
  const h16* wrrow = Wr + (size_t)ecol * CHN;
  const h16* wirow = Wi + (size_t)ecol * CHN;

  v8f accR = {}, accI = {};
#pragma unroll
  for (int kb = 0; kb < 4; ++kb) {
    int ka = kb * 32 + d0;
    v16h a_r = ld_2x8(arow + ka, arow + ka + 16);
    v16h a_i = ld_2x8(airow + ka, airow + ka + 16);
    v16h zz = {};
    v16h a_in = zz - a_i;               // for the -Ai*Bi term (packed negate)
    int kq = kb * 32 + bko;
    v16h b_r = ld_2x8(wrrow + kq, wrrow + kq + 16);
    v16h b_i = ld_2x8(wirow + kq, wirow + kq + 16);
    accR = wmma16(a_r, b_r, accR);
    accR = wmma16(a_in, b_i, accR);
    accI = wmma16(a_r, b_i, accI);
    accI = wmma16(a_i, b_r, accI);
  }

  if (mode == 0) {
    int e = nbase + col;                // 0..383 (uniform sel per tile)
    int sel = e >> 7;
    int cc = e & 127, hh = cc >> 5, dd = cc & 31;
    h16* dstR = outR0 + (size_t)sel * (2 * (size_t)QSZ);
    h16* dstI = outI0 + (size_t)sel * (2 * (size_t)QSZ);
    float br = biasR[e], bi = biasI[e];
#pragma unroll
    for (int m = 0; m < 8; ++m) {
      int row = mbase + m + rowoff;
      int bb = row >> 12, ss = row & (SPATIAL - 1);
      size_t di = (((size_t)(bb * NHEAD + hh)) * SPATIAL + ss) * DHEAD + dd;
      dstR[di] = (h16)(accR[m] + br);
      dstI[di] = (h16)(accI[m] + bi);
    }
  } else if (mode == 1) {
    int cc = nbase + col;
    float br = biasR[cc], bi = biasI[cc];
#pragma unroll
    for (int m = 0; m < 8; ++m) {
      int row = mbase + m + rowoff;
      outR0[(size_t)row * CHN + cc] = (h16)(accR[m] + br);
      outI0[(size_t)row * CHN + cc] = (h16)(accI[m] + bi);
    }
  } else {
    int cc = nbase + col;
    float br = biasR[cc], bi = biasI[cc];
#pragma unroll
    for (int m = 0; m < 8; ++m) {
      int row = mbase + m + rowoff;
      int bb = row >> 12, ss = row & (SPATIAL - 1);
      size_t idx = ((size_t)(bb * CHN + cc)) * SPATIAL + ss;
      // Output is written once and never re-read: non-temporal stores keep
      // L2 for the residual/x planes that are still live.
      __builtin_nontemporal_store(xr[idx] + accR[m] + br, &outF[idx]);
      __builtin_nontemporal_store(xi[idx] + accI[m] + bi,
                                  &outF[(size_t)BATCH * CHN * SPATIAL + idx]);
    }
  }
}

// ---------------------------------------------------------------------------
// Flash attention. Grid: (S/64, B*NH), block 128 (4 waves, 16 queries each).
// scores = (Qr Kr^T + Qi Ki^T) * scale ; softmax ; O = P Vr / P Vi.
// scale folded into the Q fragments. K tiles staged natural-layout in LDS
// (B frag reads rows of K), V tiles staged transposed (B frag reads columns
// of V). P round-trips C->A layout through a per-wave LDS scratch tile.
// Next key-block is prefetched (global_prefetch_b8) behind the current one.
// ---------------------------------------------------------------------------
__global__ __launch_bounds__(128) void attn_kernel(
    const h16* __restrict__ Qr, const h16* __restrict__ Qi,
    const h16* __restrict__ Kr, const h16* __restrict__ Ki,
    const h16* __restrict__ Vr, const h16* __restrict__ Vi,
    h16* __restrict__ Or_, h16* __restrict__ Oi_) {
  __shared__ __align__(16) h16 sKr[32][32];
  __shared__ __align__(16) h16 sKi[32][32];
  __shared__ __align__(16) h16 sVr[32][32];   // transposed: [d][key]
  __shared__ __align__(16) h16 sVi[32][32];   // transposed: [d][key]
  __shared__ __align__(16) h16 sP[4][16][32]; // per-wave P tile [row][key]

  int tid = threadIdx.x, wave = tid >> 5, lane = tid & 31;
  int bh = blockIdx.y;                        // 0..7 = b*NHEAD+h
  int qbase = blockIdx.x * 64 + wave * 16;

  size_t plane = (size_t)bh * SPATIAL * DHEAD;
  const h16* q_r = Qr + plane; const h16* q_i = Qi + plane;
  const h16* k_r = Kr + plane; const h16* k_i = Ki + plane;
  const h16* v_r = Vr + plane; const h16* v_i = Vi + plane;

  // Q fragments (A layout, K = DHEAD = 32, loaded once, pre-scaled).
  int mrow = qbase + (lane & 15);
  int d0 = (lane < 16) ? 0 : 8;
  v16h aq_r = ld_2x8(q_r + (size_t)mrow * DHEAD + d0,
                     q_r + (size_t)mrow * DHEAD + d0 + 16);
  v16h aq_i = ld_2x8(q_i + (size_t)mrow * DHEAD + d0,
                     q_i + (size_t)mrow * DHEAD + d0 + 16);
  const _Float16 hscale = (_Float16)0.17677669529663687f;  // 1/sqrt(32)
  aq_r = aq_r * hscale;
  aq_i = aq_i * hscale;

  float mstate[8], lstate[8];
#pragma unroll
  for (int m = 0; m < 8; ++m) { mstate[m] = -1e30f; lstate[m] = 0.f; }
  v8f o_r0 = {}, o_r1 = {}, o_i0 = {}, o_i1 = {};

  int col = lane & 15;
  int bko = (lane < 16) ? 0 : 8;
  int key = tid >> 2, part = tid & 3;

  for (int kb = 0; kb < SPATIAL / 32; ++kb) {
    // --- cooperative stage of 32 keys into LDS ---
    {
      size_t g = (size_t)(kb * 32 + key) * DHEAD + part * 8;
      *(uint4*)&sKr[key][part * 8] = *(const uint4*)(k_r + g);
      *(uint4*)&sKi[key][part * 8] = *(const uint4*)(k_i + g);
      union { uint4 u; h16 h[8]; } vv_r, vv_i;
      vv_r.u = *(const uint4*)(v_r + g);
      vv_i.u = *(const uint4*)(v_i + g);
#pragma unroll
      for (int j = 0; j < 8; ++j) {
        sVr[part * 8 + j][key] = vv_r.h[j];
        sVi[part * 8 + j][key] = vv_i.h[j];
      }
      // prefetch next key block into cache behind this block's compute
      if (kb + 1 < SPATIAL / 32) {
        size_t gn = g + 32 * DHEAD;
        __builtin_prefetch(k_r + gn, 0, 1);
        __builtin_prefetch(k_i + gn, 0, 1);
        __builtin_prefetch(v_r + gn, 0, 1);
        __builtin_prefetch(v_i + gn, 0, 1);
      }
    }
    __syncthreads();

    // --- scores: two 16x16 tiles (keys 0..15 and 16..31 of this block) ---
    v8f s0 = {}, s1 = {};
    {
      v16h bk_r0 = ld_2x8(&sKr[col][bko], &sKr[col][bko + 16]);
      v16h bk_i0 = ld_2x8(&sKi[col][bko], &sKi[col][bko + 16]);
      v16h bk_r1 = ld_2x8(&sKr[16 + col][bko], &sKr[16 + col][bko + 16]);
      v16h bk_i1 = ld_2x8(&sKi[16 + col][bko], &sKi[16 + col][bko + 16]);
      s0 = wmma16(aq_r, bk_r0, s0);
      s0 = wmma16(aq_i, bk_i0, s0);
      s1 = wmma16(aq_r, bk_r1, s1);
      s1 = wmma16(aq_i, bk_i1, s1);
    }

    // --- online softmax; each VGPR slot m is one query row per 16-lane half ---
    float corr[8];
#pragma unroll
    for (int m = 0; m < 8; ++m) {
      float a = s0[m], b2 = s1[m];
      float t = fmaxf(a, b2);
#pragma unroll
      for (int off = 1; off < 16; off <<= 1) t = fmaxf(t, __shfl_xor(t, off, 32));
      float mnew = fmaxf(mstate[m], t);
      float c = __expf(mstate[m] - mnew);
      float p0 = __expf(a - mnew), p1 = __expf(b2 - mnew);
      float ls = p0 + p1;
#pragma unroll
      for (int off = 1; off < 16; off <<= 1) ls += __shfl_xor(ls, off, 32);
      lstate[m] = lstate[m] * c + ls;
      mstate[m] = mnew;
      corr[m] = c;
      int prow = m + ((lane < 16) ? 0 : 8);
      sP[wave][prow][col]      = (h16)p0;
      sP[wave][prow][col + 16] = (h16)p1;
    }
#pragma unroll
    for (int m = 0; m < 8; ++m) {
      o_r0[m] *= corr[m]; o_r1[m] *= corr[m];
      o_i0[m] *= corr[m]; o_i1[m] *= corr[m];
    }
    __syncthreads();   // order sP stores before A-layout reads (and idle waves)

    // --- P (A layout) and V (B layout from transposed tile) ---
    v16h ap = ld_2x8(&sP[wave][lane & 15][d0], &sP[wave][lane & 15][d0 + 16]);
    v16h bv_r0 = ld_2x8(&sVr[col][bko], &sVr[col][bko + 16]);
    v16h bv_r1 = ld_2x8(&sVr[16 + col][bko], &sVr[16 + col][bko + 16]);
    v16h bv_i0 = ld_2x8(&sVi[col][bko], &sVi[col][bko + 16]);
    v16h bv_i1 = ld_2x8(&sVi[16 + col][bko], &sVi[16 + col][bko + 16]);
    o_r0 = wmma16(ap, bv_r0, o_r0);
    o_r1 = wmma16(ap, bv_r1, o_r1);
    o_i0 = wmma16(ap, bv_i0, o_i0);
    o_i1 = wmma16(ap, bv_i1, o_i1);
    __syncthreads();   // protect K/V tiles before next stage
  }

  // --- normalize and store concat-head layout [b*S+s, CHN] ---
  int bb = bh >> 2, hh = bh & 3;
#pragma unroll
  for (int m = 0; m < 8; ++m) {
    int row = qbase + m + ((lane < 16) ? 0 : 8);
    float invl = 1.0f / lstate[m];
    size_t base = (((size_t)bb * SPATIAL + row) * CHN) + hh * DHEAD;
    Or_[base + col]      = (h16)(o_r0[m] * invl);
    Or_[base + col + 16] = (h16)(o_r1[m] * invl);
    Oi_[base + col]      = (h16)(o_i0[m] * invl);
    Oi_[base + col + 16] = (h16)(o_i1[m] * invl);
  }
}

// ---------------------------------------------------------------------------
extern "C" void kernel_launch(void* const* d_in, const int* in_sizes, int n_in,
                              void* d_out, int out_size, void* d_ws,
                              size_t ws_size, hipStream_t stream) {
  (void)in_sizes; (void)n_in; (void)out_size; (void)ws_size;
  const float* xr      = (const float*)d_in[0];
  const float* xi      = (const float*)d_in[1];
  const float* ln_w_r  = (const float*)d_in[2];
  const float* ln_w_i  = (const float*)d_in[3];
  const float* ln_b_r  = (const float*)d_in[4];
  const float* ln_b_i  = (const float*)d_in[5];
  const float* in_w_r  = (const float*)d_in[6];
  const float* in_w_i  = (const float*)d_in[7];
  const float* in_b_r  = (const float*)d_in[8];
  const float* in_b_i  = (const float*)d_in[9];
  const float* out_w_r = (const float*)d_in[10];
  const float* out_w_i = (const float*)d_in[11];
  const float* out_b_r = (const float*)d_in[12];
  const float* out_b_i = (const float*)d_in[13];
  const float* cv_w_r  = (const float*)d_in[14];
  const float* cv_w_i  = (const float*)d_in[15];
  const float* cv_b_r  = (const float*)d_in[16];
  const float* cv_b_i  = (const float*)d_in[17];
  float* out = (float*)d_out;

  // Workspace layout (h16 elements).
  h16* ws = (h16*)d_ws;
  const size_t YN = (size_t)MROWS * CHN;   // 1,048,576
  h16* ynR = ws;
  h16* ynI = ynR + YN;
  h16* Qr = ws + 2 * YN;                   // Qr,Qi,Kr,Ki,Vr,Vi each QSZ
  h16* Qi = Qr + QSZ;
  h16* Kr = Qr + 2 * (size_t)QSZ;
  h16* Ki = Kr + QSZ;
  h16* Vr = Qr + 4 * (size_t)QSZ;
  h16* Vi = Vr + QSZ;
  h16* wp = ws + 2 * YN + 6 * (size_t)QSZ;
  h16* wInR = wp;            wp += 3 * CHN * CHN;
  h16* wInI = wp;            wp += 3 * CHN * CHN;
  h16* wOutR = wp;           wp += CHN * CHN;
  h16* wOutI = wp;           wp += CHN * CHN;
  h16* wCvR = wp;            wp += CHN * CHN;
  h16* wCvI = wp;            wp += CHN * CHN;
  h16* AOr = ynR;  h16* AOi = ynI;         // reuse: yn dead after QKV gemm
  h16* TPr = Qr;   h16* TPi = Qi;          // reuse: Q dead after attention

  // 1) LayerNorm
  ln_kernel<<<MROWS / 8, 256, 0, stream>>>(xr, xi, ln_w_r, ln_w_i, ln_b_r,
                                           ln_b_i, ynR, ynI);
  // 2) Weight conversion
  const int NIN = 3 * CHN * CHN, NSQ = CHN * CHN;
  cvt_kernel<<<(NIN + 255) / 256, 256, 0, stream>>>(in_w_r, wInR, NIN);
  cvt_kernel<<<(NIN + 255) / 256, 256, 0, stream>>>(in_w_i, wInI, NIN);
  cvt_kernel<<<(NSQ + 255) / 256, 256, 0, stream>>>(out_w_r, wOutR, NSQ);
  cvt_kernel<<<(NSQ + 255) / 256, 256, 0, stream>>>(out_w_i, wOutI, NSQ);
  cvt_kernel<<<(NSQ + 255) / 256, 256, 0, stream>>>(cv_w_r, wCvR, NSQ);
  cvt_kernel<<<(NSQ + 255) / 256, 256, 0, stream>>>(cv_w_i, wCvI, NSQ);
  // 3) QKV projection (N = 384)
  cgemm_kernel<<<dim3(3 * CHN / 16, MROWS / 64), 128, 0, stream>>>(
      ynR, ynI, wInR, wInI, in_b_r, in_b_i, 0, Qr, Qi, nullptr, nullptr,
      nullptr);
  // 4) Flash attention
  attn_kernel<<<dim3(SPATIAL / 64, BATCH * NHEAD), 128, 0, stream>>>(
      Qr, Qi, Kr, Ki, Vr, Vi, AOr, AOi);
  // 5) Out-projection (N = 128)
  cgemm_kernel<<<dim3(CHN / 16, MROWS / 64), 128, 0, stream>>>(
      AOr, AOi, wOutR, wOutI, out_b_r, out_b_i, 1, TPr, TPi, nullptr, nullptr,
      nullptr);
  // 6) 1x1 conv + residual -> d_out
  cgemm_kernel<<<dim3(CHN / 16, MROWS / 64), 128, 0, stream>>>(
      TPr, TPi, wCvR, wCvI, cv_b_r, cv_b_i, 2, nullptr, nullptr, out, xr, xi);
}
